// JetMoEFFN_20529943675425
// MI455X (gfx1250) — matrix operand order repaired and checked by reading, not compiled, not run on hardware
//
#include <hip/hip_runtime.h>
#include <hip/hip_bf16.h>

typedef __bf16 bf16_t;
typedef __bf16 bf4v  __attribute__((ext_vector_type(4)));
typedef __bf16 bf8v  __attribute__((ext_vector_type(8)));
typedef __bf16 bf16v __attribute__((ext_vector_type(16)));
typedef float  f8v   __attribute__((ext_vector_type(8)));
typedef int    v4i   __attribute__((vector_size(16)));   // async-to-LDS builtin ptr type

#if __has_builtin(__builtin_amdgcn_global_load_async_to_lds_b128) && \
    __has_builtin(__builtin_amdgcn_s_wait_asynccnt)
#define HAVE_ASYNC_LDS 1
#else
#define HAVE_ASYNC_LDS 0
#endif

// Problem sizes from setup_inputs(): B=2,S=2048,D=1024,F=2048,E=8, top_k=2
constexpr int Dd   = 1024;
constexpr int Ff   = 2048;
constexpr int Ee   = 8;
constexpr int Tt   = 4096;          // B*S
constexpr int TOPK = 2;
constexpr int TwoF = 2 * Ff;
constexpr int MT   = 64;            // token tile (4 m-subtiles of 16)
constexpr int TILES = Tt / MT;      // worst-case token tiles per expert

// ---------------- kernel 0: init out with bias, zero accumulators ----------
__global__ void moe_init_kernel(float* __restrict__ out, const float* __restrict__ bias,
                                int* counts, float* sumSel, float* sumP) {
  int idx = blockIdx.x * blockDim.x + threadIdx.x;
  if (idx < Tt * Dd) out[idx] = bias[idx & (Dd - 1)];
  if (blockIdx.x == 0 && threadIdx.x < Ee) {
    counts[threadIdx.x] = 0;
    sumSel[threadIdx.x] = 0.f;
    sumP[threadIdx.x]   = 0.f;
  }
}

// ---------------- kernel 1: router (one wave32 per token) ------------------
__global__ __launch_bounds__(256) void moe_router_kernel(
    const float* __restrict__ x, const float* __restrict__ rw,
    int* counts, int* slot_list, float* gate_list,
    float* sumSel, float* sumP) {
  int wave = threadIdx.x >> 5;
  int lane = threadIdx.x & 31;
  int t = blockIdx.x * 8 + wave;
  if (t >= Tt) return;

  float acc[Ee];
#pragma unroll
  for (int e = 0; e < Ee; e++) acc[e] = 0.f;
  for (int d = lane; d < Dd; d += 32) {
    float xv = x[(size_t)t * Dd + d];
    const float* r = rw + (size_t)d * Ee;
#pragma unroll
    for (int e = 0; e < Ee; e++) acc[e] += xv * r[e];
  }
#pragma unroll
  for (int e = 0; e < Ee; e++) {
#pragma unroll
    for (int off = 16; off; off >>= 1) acc[e] += __shfl_xor(acc[e], off, 32);
  }
  if (lane == 0) {
    int i0 = 0;
    for (int e = 1; e < Ee; e++) if (acc[e] > acc[i0]) i0 = e;
    int i1 = (i0 == 0) ? 1 : 0;
    for (int e = 0; e < Ee; e++) { if (e == i0) continue; if (acc[e] > acc[i1]) i1 = e; }
    float m  = fmaxf(acc[i0], acc[i1]);
    float e0 = __expf(acc[i0] - m), e1 = __expf(acc[i1] - m);
    float inv = 1.f / (e0 + e1);
    int p0 = atomicAdd(&counts[i0], 1);
    slot_list[i0 * Tt + p0] = t * 2 + 0;
    gate_list[i0 * Tt + p0] = e0 * inv;
    int p1 = atomicAdd(&counts[i1], 1);
    slot_list[i1 * Tt + p1] = t * 2 + 1;
    gate_list[i1 * Tt + p1] = e1 * inv;
    atomicAdd(&sumSel[i0], 1.f);
    atomicAdd(&sumSel[i1], 1.f);
    float mx = acc[0];
    for (int e = 1; e < Ee; e++) mx = fmaxf(mx, acc[e]);
    float s = 0.f, ex[Ee];
    for (int e = 0; e < Ee; e++) { ex[e] = __expf(acc[e] - mx); s += ex[e]; }
    float invs = 1.f / s;
    for (int e = 0; e < Ee; e++) atomicAdd(&sumP[e], ex[e] * invs);
  }
}

// WMMA fragment helpers (layouts per CDNA5 ISA 7.12.2, wave32)
__device__ inline bf16v load_a_frag(const bf16_t* row, int half) {
  // A 16x32 bf16: lane m holds K runs [half*8, +8) and [16+half*8, +8)
  bf8v lo = *(const bf8v*)(row + half * 8);
  bf8v hi = *(const bf8v*)(row + 16 + half * 8);
  return __builtin_shufflevector(lo, hi, 0, 1, 2, 3, 4, 5, 6, 7,
                                 8, 9, 10, 11, 12, 13, 14, 15);
}
__device__ inline bf16v load_b_frag(const bf16_t* row, int half) {
  // B 32x16 bf16: lane (n + 16*half) holds K runs [half*8, +8) and [16+half*8, +8)
  bf8v lo = *(const bf8v*)(row + half * 8);
  bf8v hi = *(const bf8v*)(row + 16 + half * 8);
  return __builtin_shufflevector(lo, hi, 0, 1, 2, 3, 4, 5, 6, 7,
                                 8, 9, 10, 11, 12, 13, 14, 15);
}

#define WMMA_BF16(A, B, C, RA) \
  __builtin_amdgcn_wmma_f32_16x16x32_bf16(false, (A), false, (B), (short)0, (C), (RA), false)

// ---------------- kernel 2a: grouped GEMM x @ w_in[e], GLU, gate -----------
// block = 256 threads (8 waves). Block tile: 64 tokens x 64 F cols (g and v).
// wave w: m-subtile (w&3), n-subtiles {2*(w>>2), 2*(w>>2)+1}.
__global__ __launch_bounds__(256) void moe_ffn_in_kernel(
    const float* __restrict__ x, const float* __restrict__ w_in,
    const int* __restrict__ counts, const int* __restrict__ slot_list,
    const float* __restrict__ gate_list, bf16_t* __restrict__ act) {
  int e    = blockIdx.x / TILES;
  int tile = blockIdx.x % TILES;
  int cnt  = counts[e];
  if (tile * MT >= cnt) return;           // uniform early exit
  int fbase = blockIdx.y * 64;
  const float* W = w_in + (size_t)e * Dd * TwoF;

  __shared__ int    s_slot[MT];
  __shared__ float  s_gate[MT];
  __shared__ __align__(32) bf16_t s_x[MT * 32];     // A tile, [m][k]
  __shared__ __align__(32) bf16_t s_bg[64 * 40];    // B tile g, [n][k] padded
  __shared__ __align__(32) bf16_t s_bv[64 * 40];    // B tile v

  int tid = threadIdx.x;
  if (tid < MT) {
    int p = tile * MT + tid;
    if (p < cnt) {
      s_slot[tid] = slot_list[e * Tt + p];
      s_gate[tid] = gate_list[e * Tt + p];
    } else {
      s_slot[tid] = -1;
      s_gate[tid] = 0.f;
    }
  }
  __syncthreads();

  // ---- hoisted per-thread staging state (loop-invariant) ----
  int xm0 = tid >> 3;                 // row for s=0 (i = tid*4, m = i>>5)
  int xm1 = xm0 + 32;                 // row for s=1
  int xk  = (tid * 4) & 31;
  int slot0 = s_slot[xm0];
  int slot1 = s_slot[xm1];
  const float* xrow0 = x + (size_t)(slot0 >= 0 ? (slot0 >> 1) : 0) * Dd + xk;
  const float* xrow1 = x + (size_t)(slot1 >= 0 ? (slot1 >> 1) : 0) * Dd + xk;
  {  // pad rows stay zero for the whole kernel; fill once
    bf4v z4 = {};
    if (slot0 < 0) *(bf4v*)(s_x + xm0 * 32 + xk) = z4;
    if (slot1 < 0) *(bf4v*)(s_x + xm1 * 32 + xk) = z4;
  }
  int wn  = (tid * 4) & 63;           // weight tile (k,n) indices
  int wk0 = (tid * 4) >> 6;
  int wk1 = wk0 + 16;

  int wave = tid >> 5, lane = tid & 31;
  int half = lane >> 4, ln = lane & 15;
  int mw = wave & 3;
  int nw = (wave >> 2) * 2;

  f8v accg0 = {}, accv0 = {}, accg1 = {}, accv1 = {};

  for (int kk = 0; kk < Dd; kk += 32) {
    __syncthreads();
    const float* Wk = W + (size_t)kk * TwoF + fbase;
    // stage x tile rows (float4/b128 loads, fp32 -> bf16, b64 LDS stores)
    if (slot0 >= 0) {
      float4 v = *(const float4*)(xrow0 + kk);
      bf4v pk = {(bf16_t)v.x, (bf16_t)v.y, (bf16_t)v.z, (bf16_t)v.w};
      *(bf4v*)(s_x + xm0 * 32 + xk) = pk;
    }
    if (slot1 >= 0) {
      float4 v = *(const float4*)(xrow1 + kk);
      bf4v pk = {(bf16_t)v.x, (bf16_t)v.y, (bf16_t)v.z, (bf16_t)v.w};
      *(bf4v*)(s_x + xm1 * 32 + xk) = pk;
    }
    // stage w_in tiles [32k x 64n], coalesced float4 along n, both g and v
#pragma unroll
    for (int s = 0; s < 2; s++) {
      int k = s ? wk1 : wk0;
      float4 g4 = *(const float4*)(Wk + (size_t)k * TwoF + wn);
      float4 v4 = *(const float4*)(Wk + (size_t)k * TwoF + wn + Ff);
      s_bg[(wn + 0) * 40 + k] = (bf16_t)g4.x;
      s_bg[(wn + 1) * 40 + k] = (bf16_t)g4.y;
      s_bg[(wn + 2) * 40 + k] = (bf16_t)g4.z;
      s_bg[(wn + 3) * 40 + k] = (bf16_t)g4.w;
      s_bv[(wn + 0) * 40 + k] = (bf16_t)v4.x;
      s_bv[(wn + 1) * 40 + k] = (bf16_t)v4.y;
      s_bv[(wn + 2) * 40 + k] = (bf16_t)v4.z;
      s_bv[(wn + 3) * 40 + k] = (bf16_t)v4.w;
    }
    __syncthreads();
    if (kk + 32 < Dd)   // global_prefetch_b8 for the next K tile
      __builtin_prefetch(Wk + (size_t)(32 + (tid >> 3)) * TwoF + (tid & 7) * 8, 0, 0);

    bf16v a   = load_a_frag(s_x + (mw * 16 + ln) * 32, half);
    bf16v bg0 = load_b_frag(s_bg + ((nw + 0) * 16 + ln) * 40, half);
    bf16v bv0 = load_b_frag(s_bv + ((nw + 0) * 16 + ln) * 40, half);
    bf16v bg1 = load_b_frag(s_bg + ((nw + 1) * 16 + ln) * 40, half);
    bf16v bv1 = load_b_frag(s_bv + ((nw + 1) * 16 + ln) * 40, half);
    accg0 = WMMA_BF16(a, bg0, accg0, false);
    accv0 = WMMA_BF16(a, bv0, accv0, true);   // same A as previous -> RA hint
    accg1 = WMMA_BF16(a, bg1, accg1, true);
    accv1 = WMMA_BF16(a, bv1, accv1, true);
  }

  // epilogue: act = silu(g) * v * gate
#pragma unroll
  for (int j = 0; j < 2; j++) {
    const f8v& G = j ? accg1 : accg0;
    const f8v& V = j ? accv1 : accv0;
#pragma unroll
    for (int r = 0; r < 8; r++) {
      int m = mw * 16 + r + half * 8;          // C layout: row = r + 8*half
      int slot = s_slot[m];
      if (slot >= 0) {
        float g = G[r], v = V[r];
        float av = g / (1.f + __expf(-g)) * v * s_gate[m];
        act[(size_t)slot * Ff + fbase + (nw + j) * 16 + ln] = (bf16_t)av;
      }
    }
  }
}

// ---------------- kernel 2b: grouped GEMM act @ w_out[e] -> scatter add ----
// Block tile: 64 tokens x 64 D cols. wave w: m (w&3), n {2*(w>>2), +1}.
__global__ __launch_bounds__(256) void moe_ffn_out_kernel(
    const bf16_t* __restrict__ act, const float* __restrict__ w_out,
    const int* __restrict__ counts, const int* __restrict__ slot_list,
    float* __restrict__ out) {
  int e    = blockIdx.x / TILES;
  int tile = blockIdx.x % TILES;
  int cnt  = counts[e];
  if (tile * MT >= cnt) return;
  int dbase = blockIdx.y * 64;
  const float* W = w_out + (size_t)e * Ff * Dd;

  __shared__ int s_slot[MT];
  __shared__ __align__(32) bf16_t s_a[MT * 32];
  __shared__ __align__(32) bf16_t s_b[64 * 40];

  int tid = threadIdx.x;
  if (tid < MT) {
    int p = tile * MT + tid;
    s_slot[tid] = (p < cnt) ? slot_list[e * Tt + p] : -1;
  }
  __syncthreads();

  // ---- hoisted per-thread staging state ----
  int am = tid >> 2;                  // i = tid*8, m = i>>5
  int ak = (tid * 8) & 31;
  int aslot = s_slot[am];
  const bf16_t* arow = act + (size_t)(aslot >= 0 ? aslot : 0) * Ff + ak;
  {  // pad rows zero once
    bf8v z = {};
    if (aslot < 0) *(bf8v*)(s_a + am * 32 + ak) = z;
  }
  int wn  = (tid * 4) & 63;
  int wk0 = (tid * 4) >> 6;
  int wk1 = wk0 + 16;

  int wave = tid >> 5, lane = tid & 31;
  int half = lane >> 4, ln = lane & 15;
  int mw = wave & 3;
  int nw = (wave >> 2) * 2;
  f8v acc0 = {}, acc1 = {};

  for (int kk = 0; kk < Ff; kk += 32) {
    __syncthreads();
    const float* Wk = W + (size_t)kk * Dd + dbase;
    // stage act tile row slice (one b128 per thread, async to LDS)
    if (aslot >= 0) {
#if HAVE_ASYNC_LDS
      __builtin_amdgcn_global_load_async_to_lds_b128(
          (v4i*)(arow + kk), (v4i*)(s_a + am * 32 + ak), 0, 0);
#else
      *(bf8v*)(s_a + am * 32 + ak) = *(const bf8v*)(arow + kk);
#endif
    }
    // stage w_out tile [32k x 64n], coalesced float4 along n
#pragma unroll
    for (int s = 0; s < 2; s++) {
      int k = s ? wk1 : wk0;
      float4 w4 = *(const float4*)(Wk + (size_t)k * Dd + wn);
      s_b[(wn + 0) * 40 + k] = (bf16_t)w4.x;
      s_b[(wn + 1) * 40 + k] = (bf16_t)w4.y;
      s_b[(wn + 2) * 40 + k] = (bf16_t)w4.z;
      s_b[(wn + 3) * 40 + k] = (bf16_t)w4.w;
    }
#if HAVE_ASYNC_LDS
    __builtin_amdgcn_s_wait_asynccnt(0);
#endif
    __syncthreads();
    if (kk + 32 < Ff)
      __builtin_prefetch(Wk + (size_t)(32 + (tid >> 3)) * Dd + (tid & 7) * 8, 0, 0);

    bf16v a  = load_a_frag(s_a + (mw * 16 + ln) * 32, half);
    bf16v b0 = load_b_frag(s_b + ((nw + 0) * 16 + ln) * 40, half);
    bf16v b1 = load_b_frag(s_b + ((nw + 1) * 16 + ln) * 40, half);
    acc0 = WMMA_BF16(a, b0, acc0, false);
    acc1 = WMMA_BF16(a, b1, acc1, true);      // same A -> RA hint
  }

#pragma unroll
  for (int j = 0; j < 2; j++) {
    const f8v& C = j ? acc1 : acc0;
#pragma unroll
    for (int r = 0; r < 8; r++) {
      int m = mw * 16 + r + half * 8;
      int slot = s_slot[m];
      if (slot >= 0)
        atomicAdd(&out[(size_t)(slot >> 1) * Dd + dbase + (nw + j) * 16 + ln], C[r]);
    }
  }
}

// ---------------- kernel 3: finalize aux loss ------------------------------
__global__ void moe_aux_kernel(const float* sumSel, const float* sumP, float* out_aux) {
  if (threadIdx.x == 0 && blockIdx.x == 0) {
    float a = 0.f;
    for (int e = 0; e < Ee; e++)
      a += (sumSel[e] / (float)(Tt * TOPK)) * (sumP[e] / (float)Tt);
    *out_aux = (float)Ee * a;
  }
}

extern "C" void kernel_launch(void* const* d_in, const int* in_sizes, int n_in,
                              void* d_out, int out_size, void* d_ws, size_t ws_size,
                              hipStream_t stream) {
  const float* x     = (const float*)d_in[0];
  const float* rw    = (const float*)d_in[1];
  const float* w_in  = (const float*)d_in[2];
  const float* w_out = (const float*)d_in[3];
  const float* bias  = (const float*)d_in[4];
  float* out = (float*)d_out;                 // [T*D] out + [1] aux

  char* ws = (char*)d_ws;
  int*    counts    = (int*)ws;                                  // 8
  float*  sumSel    = (float*)(ws + 64);                         // 8
  float*  sumP      = (float*)(ws + 128);                        // 8
  int*    slot_list = (int*)(ws + 256);                          // E*T
  float*  gate_list = (float*)(ws + 256 + (size_t)Ee * Tt * 4);  // E*T
  bf16_t* act       = (bf16_t*)(ws + 256 + (size_t)Ee * Tt * 8); // 2T*F bf16

  moe_init_kernel<<<(Tt * Dd + 255) / 256, 256, 0, stream>>>(out, bias, counts, sumSel, sumP);
  moe_router_kernel<<<Tt / 8, 256, 0, stream>>>(x, rw, counts, slot_list, gate_list, sumSel, sumP);
  moe_ffn_in_kernel<<<dim3(Ee * TILES, Ff / 64), 256, 0, stream>>>(
      x, w_in, counts, slot_list, gate_list, act);
  moe_ffn_out_kernel<<<dim3(Ee * TILES, Dd / 64), 256, 0, stream>>>(
      act, w_out, counts, slot_list, out);
  moe_aux_kernel<<<1, 32, 0, stream>>>(sumSel, sumP, out + (size_t)Tt * Dd);
}
